// GNN_22110491639893
// MI455X (gfx1250) — matrix operand order repaired and checked by reading, not compiled, not run on hardware
//
#include <hip/hip_runtime.h>
#include <hip/hip_bf16.h>

#define NUM_LAYER 5
#define EMB 300
#define HID 600
#define NNODES 100000
#define NEDGES 250000
#define BN_EPS 1e-5f

// padded K dims (multiple of 32) and padded col dims (multiple of 64, 4 tiles/strip)
#define KP1 320   // EMB padded      (GEMM1 K)
#define KP2 608   // HID padded      (GEMM2 K, also GEMM1 output stride)
#define NP1 640   // HID col-padded  (GEMM1 W rows: 10 * 64)
#define NP2 320   // EMB col-padded  (GEMM2 W rows: 5 * 64)
#define LDS_PAD 8 // +8 bf16 (=16B) row padding to spread LDS banks

typedef __attribute__((ext_vector_type(8)))  __bf16 v8bf;
typedef __attribute__((ext_vector_type(16))) __bf16 v16bf;
typedef __attribute__((ext_vector_type(8)))  float  v8f;

__device__ __forceinline__ v16bf cat8(v8bf lo, v8bf hi) {
    return __builtin_shufflevector(lo, hi, 0,1,2,3,4,5,6,7,8,9,10,11,12,13,14,15);
}

// ---------------- init: h = x_emb1[x[:,0]] + x_emb2[x[:,1]] ----------------
__global__ void k_init_h(const int* __restrict__ x,
                         const float* __restrict__ e1,
                         const float* __restrict__ e2,
                         float* __restrict__ h, long long total) {
    long long t = (long long)blockIdx.x * blockDim.x + threadIdx.x;
    if (t >= total) return;
    int node = (int)(t / EMB);
    int d    = (int)(t % EMB);
    int a = x[node * 2 + 0];
    int b = x[node * 2 + 1];
    h[t] = e1[(long long)a * EMB + d] + e2[(long long)b * EMB + d];
}

// ---------------- zero a float buffer ----------------
__global__ void k_zero(float* __restrict__ p, long long total) {
    long long t = (long long)blockIdx.x * blockDim.x + threadIdx.x;
    if (t < total) p[t] = 0.0f;
}

// -------- fused edge embedding + gather h[src] + atomic scatter-add to agg --------
__global__ void k_scatter(const float* __restrict__ h,
                          const int* __restrict__ ei,    // [2, NEDGES]
                          const int* __restrict__ ea,    // [NEDGES, 2]
                          const float* __restrict__ ee1, // layer slice [6, EMB]
                          const float* __restrict__ ee2, // layer slice [3, EMB]
                          float* __restrict__ agg, long long total) {
    long long t = (long long)blockIdx.x * blockDim.x + threadIdx.x;
    if (t >= total) return;
    int e = (int)(t / EMB);
    int d = (int)(t % EMB);
    int src, dst, a0, a1;
    if (e < NEDGES) {
        src = ei[e];
        dst = ei[NEDGES + e];
        a0  = ea[e * 2 + 0];
        a1  = ea[e * 2 + 1];
    } else {           // self loops: edge_attr = [4, 0]
        src = dst = e - NEDGES;
        a0 = 4; a1 = 0;
    }
    float v = h[(long long)src * EMB + d] + ee1[a0 * EMB + d] + ee2[a1 * EMB + d];
    unsafeAtomicAdd(&agg[(long long)dst * EMB + d], v);  // L2-resident scatter-add
}

// -------- convert f32 [rows,K] -> zero-padded bf16 [rows_p,Kp] --------
__global__ void k_cvt_pad(const float* __restrict__ in, __bf16* __restrict__ out,
                          int rows, int K, int rows_p, int Kp) {
    long long t = (long long)blockIdx.x * blockDim.x + threadIdx.x;
    long long total = (long long)rows_p * Kp;
    if (t >= total) return;
    int r = (int)(t / Kp);
    int c = (int)(t % Kp);
    float v = (r < rows && c < K) ? in[(long long)r * K + c] : 0.0f;
    out[t] = (__bf16)v;
}

// -------- WMMA bf16 GEMM: C[M,N] = act(A[M,Kp] @ Wp[Np,Kp]^T + bias) --------
// Block: 256 threads = 8 waves stacked on M; each wave computes a 32x64 strip
// (2 row-tiles x 4 col-tiles = 8 WMMA accumulators). The block's 64-column
// B strip (64 x Kp bf16) is staged once in LDS (bank-padded) and shared by
// all 8 waves. Grid: (colStrips, ceil(Mtiles32/8)). Requires M % 32 == 0.
__global__ void k_gemm(const __bf16* __restrict__ A,
                       const __bf16* __restrict__ Wp,
                       const float* __restrict__ bias,
                       void* __restrict__ Cout, int ldo, int outBf16,
                       int M, int N, int Kp, int relu) {
    extern __shared__ __bf16 smem[];          // [64][Kp + LDS_PAD]
    const int lds = Kp + LDS_PAD;
    const int col0 = blockIdx.x << 6;

    // ---- cooperative B-strip load: 64 columns x Kp (16B vector chunks) ----
    {
        int kc = Kp >> 3;                     // v8bf chunks per column
        int chunks = kc << 6;                 // 64 * kc
        for (int q = threadIdx.x; q < chunks; q += blockDim.x) {
            int c = q / kc;
            int j = q - c * kc;
            *(v8bf*)(smem + c * lds + (j << 3)) =
                *(const v8bf*)(Wp + (long long)(col0 + c) * Kp + (j << 3));
        }
    }
    __syncthreads();

    int wave = threadIdx.x >> 5;
    int lane = threadIdx.x & 31;
    int row0 = (blockIdx.y * 8 + wave) << 5;  // 32 rows per wave
    if (row0 >= M) return;                    // after the only barrier: safe

    int grp     = lane >> 4;                  // 0 or 1
    int kbA     = grp << 4;                   // A half-wave K offset (+16)
    int kbB     = grp << 4;                   // B half-wave K offset (+16)
    int mrow    = lane & 15;
    int lanecol = lane & 15;

    const __bf16* arow0 = A + (long long)(row0 + mrow) * Kp + kbA;
    const __bf16* arow1 = A + (long long)(row0 + 16 + mrow) * Kp + kbA;
    const __bf16* bp0 = smem + (0  + lanecol) * lds + kbB;
    const __bf16* bp1 = smem + (16 + lanecol) * lds + kbB;
    const __bf16* bp2 = smem + (32 + lanecol) * lds + kbB;
    const __bf16* bp3 = smem + (48 + lanecol) * lds + kbB;

    v8f acc00 = {}, acc01 = {}, acc02 = {}, acc03 = {};
    v8f acc10 = {}, acc11 = {}, acc12 = {}, acc13 = {};

    for (int k0 = 0; k0 < Kp; k0 += 32) {
        v16bf a0 = cat8(*(const v8bf*)(arow0 + k0), *(const v8bf*)(arow0 + k0 + 8));
        v16bf a1 = cat8(*(const v8bf*)(arow1 + k0), *(const v8bf*)(arow1 + k0 + 8));
        v16bf b0 = cat8(*(const v8bf*)(bp0 + k0), *(const v8bf*)(bp0 + k0 + 8));
        v16bf b1 = cat8(*(const v8bf*)(bp1 + k0), *(const v8bf*)(bp1 + k0 + 8));
        v16bf b2 = cat8(*(const v8bf*)(bp2 + k0), *(const v8bf*)(bp2 + k0 + 8));
        v16bf b3 = cat8(*(const v8bf*)(bp3 + k0), *(const v8bf*)(bp3 + k0 + 8));
        acc00 = __builtin_amdgcn_wmma_f32_16x16x32_bf16(false, a0, false, b0, (short)0, acc00, false, false);
        acc10 = __builtin_amdgcn_wmma_f32_16x16x32_bf16(false, a1, false, b0, (short)0, acc10, false, false);
        acc01 = __builtin_amdgcn_wmma_f32_16x16x32_bf16(false, a0, false, b1, (short)0, acc01, false, false);
        acc11 = __builtin_amdgcn_wmma_f32_16x16x32_bf16(false, a1, false, b1, (short)0, acc11, false, false);
        acc02 = __builtin_amdgcn_wmma_f32_16x16x32_bf16(false, a0, false, b2, (short)0, acc02, false, false);
        acc12 = __builtin_amdgcn_wmma_f32_16x16x32_bf16(false, a1, false, b2, (short)0, acc12, false, false);
        acc03 = __builtin_amdgcn_wmma_f32_16x16x32_bf16(false, a0, false, b3, (short)0, acc03, false, false);
        acc13 = __builtin_amdgcn_wmma_f32_16x16x32_bf16(false, a1, false, b3, (short)0, acc13, false, false);
    }

    v8f accs[2][4] = {{acc00, acc01, acc02, acc03}, {acc10, acc11, acc12, acc13}};
#pragma unroll
    for (int rt = 0; rt < 2; ++rt) {
        int rbase = row0 + (rt << 4) + (grp << 3);   // D layout: VGPR v -> row v (+8 grp1)
#pragma unroll
        for (int t = 0; t < 4; ++t) {
            int col = col0 + (t << 4) + lanecol;
            if (col < N) {
                float bv = bias[col];
#pragma unroll
                for (int v = 0; v < 8; ++v) {
                    float o = accs[rt][t][v] + bv;
                    if (relu) o = fmaxf(o, 0.0f);
                    if (outBf16) ((__bf16*)Cout)[(long long)(rbase + v) * ldo + col] = (__bf16)o;
                    else         ((float*)Cout)[(long long)(rbase + v) * ldo + col] = o;
                }
            } else if (outBf16 && col < ldo) {       // keep padded bf16 output zeroed
#pragma unroll
                for (int v = 0; v < 8; ++v)
                    ((__bf16*)Cout)[(long long)(rbase + v) * ldo + col] = (__bf16)0.0f;
            }
        }
    }
}

// -------- BatchNorm stats: one block per channel --------
__global__ void k_bn_stats(const float* __restrict__ h,
                           float* __restrict__ meanv, float* __restrict__ istdv,
                           int n, int d) {
    int c = blockIdx.x;
    float s1 = 0.0f, s2 = 0.0f;
    for (int i = threadIdx.x; i < n; i += blockDim.x) {
        float v = h[(long long)i * d + c];
        s1 += v; s2 += v * v;
    }
    __shared__ float a1[256], a2[256];
    a1[threadIdx.x] = s1; a2[threadIdx.x] = s2;
    __syncthreads();
    for (int off = 128; off > 0; off >>= 1) {
        if ((int)threadIdx.x < off) {
            a1[threadIdx.x] += a1[threadIdx.x + off];
            a2[threadIdx.x] += a2[threadIdx.x + off];
        }
        __syncthreads();
    }
    if (threadIdx.x == 0) {
        float mu  = a1[0] / (float)n;
        float var = a2[0] / (float)n - mu * mu;
        meanv[c] = mu;
        istdv[c] = rsqrtf(var + BN_EPS);
    }
}

// -------- BatchNorm apply (+ optional ReLU) --------
__global__ void k_bn_apply(const float* __restrict__ in, float* __restrict__ out,
                           const float* __restrict__ meanv, const float* __restrict__ istdv,
                           const float* __restrict__ gamma, const float* __restrict__ beta,
                           long long total, int d, int relu) {
    long long t = (long long)blockIdx.x * blockDim.x + threadIdx.x;
    if (t >= total) return;
    int c = (int)(t % d);
    float v = (in[t] - meanv[c]) * istdv[c] * gamma[c] + beta[c];
    if (relu) v = fmaxf(v, 0.0f);
    out[t] = v;
}

extern "C" void kernel_launch(void* const* d_in, const int* in_sizes, int n_in,
                              void* d_out, int out_size, void* d_ws, size_t ws_size,
                              hipStream_t stream) {
    const int*   x     = (const int*)d_in[0];
    const int*   ei    = (const int*)d_in[1];
    const int*   ea    = (const int*)d_in[2];
    const float* xe1   = (const float*)d_in[3];
    const float* xe2   = (const float*)d_in[4];
    const float* ee1   = (const float*)d_in[5];
    const float* ee2   = (const float*)d_in[6];
    const float* w1    = (const float*)d_in[7];
    const float* b1    = (const float*)d_in[8];
    const float* w2    = (const float*)d_in[9];
    const float* b2    = (const float*)d_in[10];
    const float* gamma = (const float*)d_in[11];
    const float* beta  = (const float*)d_in[12];
    float* out = (float*)d_out;

    // ---- workspace layout ----
    float* h    = (float*)d_ws;                           // N*EMB f32
    float* agg  = h + (size_t)NNODES * EMB;               // N*EMB f32
    float* mu   = agg + (size_t)NNODES * EMB;             // 512 f32 (stats)
    float* istd = mu + 512;                               // 512 f32
    __bf16* abf   = (__bf16*)(istd + 512);                // N*KP1 bf16 (GEMM1 A)
    __bf16* hidbf = abf + (size_t)NNODES * KP1;           // N*KP2 bf16 (GEMM1 out / GEMM2 A)
    __bf16* w1p   = hidbf + (size_t)NNODES * KP2;         // NP1*KP1 bf16
    __bf16* w2p   = w1p + (size_t)NP1 * KP1;              // NP2*KP2 bf16

    const int B = 256;
    const long long nh   = (long long)NNODES * EMB;
    const long long nmsg = (long long)(NEDGES + NNODES) * EMB;
    const int mtiles32  = NNODES / 32;                    // 3125
    const int rowBlocks = (mtiles32 + 7) / 8;             // 391

    k_init_h<<<(unsigned)((nh + B - 1) / B), B, 0, stream>>>(x, xe1, xe2, h, nh);

    for (int l = 0; l < NUM_LAYER; ++l) {
        k_zero<<<(unsigned)((nh + B - 1) / B), B, 0, stream>>>(agg, nh);
        k_scatter<<<(unsigned)((nmsg + B - 1) / B), B, 0, stream>>>(
            h, ei, ea, ee1 + (size_t)l * 6 * EMB, ee2 + (size_t)l * 3 * EMB, agg, nmsg);

        // convert agg -> bf16 [N, KP1], and weights -> padded bf16
        {
            long long ta = (long long)NNODES * KP1;
            k_cvt_pad<<<(unsigned)((ta + B - 1) / B), B, 0, stream>>>(agg, abf, NNODES, EMB, NNODES, KP1);
            long long tw1 = (long long)NP1 * KP1;
            k_cvt_pad<<<(unsigned)((tw1 + B - 1) / B), B, 0, stream>>>(
                w1 + (size_t)l * HID * EMB, w1p, HID, EMB, NP1, KP1);
            long long tw2 = (long long)NP2 * KP2;
            k_cvt_pad<<<(unsigned)((tw2 + B - 1) / B), B, 0, stream>>>(
                w2 + (size_t)l * EMB * HID, w2p, EMB, HID, NP2, KP2);
        }

        // GEMM1: hidbf = relu(abf @ w1p^T + b1) as bf16 [N, KP2]   (M=100000, N=600, Kp=320)
        {
            dim3 grid(NP1 / 64, rowBlocks);
            size_t shmem = (size_t)64 * (KP1 + LDS_PAD) * sizeof(__bf16);
            k_gemm<<<grid, B, shmem, stream>>>(
                abf, w1p, b1 + (size_t)l * HID, (void*)hidbf, KP2, 1,
                NNODES, HID, KP1, 1);
        }
        // GEMM2: h = hidbf @ w2p^T + b2 as f32 [N, EMB]            (M=100000, N=300, Kp=608)
        {
            dim3 grid(NP2 / 64, rowBlocks);
            size_t shmem = (size_t)64 * (KP2 + LDS_PAD) * sizeof(__bf16);
            k_gemm<<<grid, B, shmem, stream>>>(
                hidbf, w2p, b2 + (size_t)l * EMB, (void*)h, EMB, 0,
                NNODES, EMB, KP2, 0);
        }

        k_bn_stats<<<EMB, 256, 0, stream>>>(h, mu, istd, NNODES, EMB);
        float* o = (l == NUM_LAYER - 1) ? out : h;
        k_bn_apply<<<(unsigned)((nh + B - 1) / B), B, 0, stream>>>(
            h, o, mu, istd, gamma + (size_t)l * EMB, beta + (size_t)l * EMB,
            nh, EMB, (l < NUM_LAYER - 1) ? 1 : 0);
    }
}
